// MoE_24343874633735
// MI455X (gfx1250) — matrix operand order repaired and checked by reading, not compiled, not run on hardware
//
#include <hip/hip_runtime.h>
#include <hip/hip_bf16.h>

#define DIMD 2048
#define NEXP 8
#define HID  1408
#define HSH  2816
#define NTOK 2048
#define MT   32          // tokens per block tile (two 16-row WMMA sub-tiles)

typedef __attribute__((ext_vector_type(16))) __bf16 v16bf;
typedef __attribute__((ext_vector_type(8)))  float  v8f;

union Frag { v16bf v; uint4 q[2]; };

__device__ __forceinline__ unsigned short f2bf(float f) {
    unsigned u = __float_as_uint(f);
    unsigned r = u + 0x7FFFu + ((u >> 16) & 1u);   // round-to-nearest-even
    return (unsigned short)(r >> 16);
}

// A fragment: 16x32 bf16 tile from LDS (row-major, rowStride in bf16 elems).
// Lane L: row m = L&15; half = L>>4. VGPR0..7 = K kbase+16*half+0..15
// (contiguous order, matching the B fragment).
__device__ __forceinline__ v16bf load_a_lds(const unsigned short* buf, int rowStride,
                                            int kbase, int lane) {
    const int half = lane >> 4, m = lane & 15;
    const unsigned short* p = buf + (size_t)m * rowStride + kbase + half * 16;
    Frag f;
    f.q[0] = *(const uint4*)(p);
    f.q[1] = *(const uint4*)(p + 8);
    return f.v;
}

// B fragment: 32x16 bf16. Lane L holds column n = L&15; lanes 0-15 K=kbase..+15,
// lanes 16-31 K=kbase+16..+31 -> 16 contiguous bf16 from a weight row.
__device__ __forceinline__ v16bf load_b_glb(const unsigned short* wrow) {
    Frag f;
    f.q[0] = *(const uint4*)(wrow);
    f.q[1] = *(const uint4*)(wrow + 8);
    return f.v;
}

__device__ __forceinline__ v8f wmma_bf16(v16bf a, v16bf b, v8f c) {
    return __builtin_amdgcn_wmma_f32_16x16x32_bf16(false, a, false, b, (short)0, c,
                                                   false, false);
}

// ---------------------------------------------------------------- utility
__global__ void moe_zero_counts(int* counts) {
    if (threadIdx.x < NEXP) counts[threadIdx.x] = 0;
}

__global__ void moe_cvt_bf16(const float4* __restrict__ s, ushort4* __restrict__ d,
                             long n4) {
    long i = (long)blockIdx.x * blockDim.x + threadIdx.x;
    long stride = (long)gridDim.x * blockDim.x;
    for (; i < n4; i += stride) {
        float4 v = s[i];
        ushort4 o;
        o.x = f2bf(v.x); o.y = f2bf(v.y); o.z = f2bf(v.z); o.w = f2bf(v.w);
        d[i] = o;
    }
}

// ---------------------------------------------------------------- gating
__global__ __launch_bounds__(256) void moe_gate(
        const float* __restrict__ x, const float* __restrict__ gw,
        int* __restrict__ counts, int* __restrict__ tokList,
        int* __restrict__ slotList, float* __restrict__ gateList) {
    const int wave = threadIdx.x >> 5, lane = threadIdx.x & 31;
    const int tok = blockIdx.x * 8 + wave;
    float acc[NEXP];
#pragma unroll
    for (int e = 0; e < NEXP; ++e) acc[e] = 0.f;
    const float* xr = x + (size_t)tok * DIMD;
    for (int k = lane; k < DIMD; k += 32) {
        float xv = xr[k];
#pragma unroll
        for (int e = 0; e < NEXP; ++e) acc[e] = fmaf(xv, gw[e * DIMD + k], acc[e]);
    }
#pragma unroll
    for (int e = 0; e < NEXP; ++e)
#pragma unroll
        for (int off = 16; off > 0; off >>= 1)
            acc[e] += __shfl_xor(acc[e], off, 32);
    if (lane == 0) {
        float mx = acc[0];
#pragma unroll
        for (int e = 1; e < NEXP; ++e) mx = fmaxf(mx, acc[e]);
        float p[NEXP], s = 0.f;
#pragma unroll
        for (int e = 0; e < NEXP; ++e) { p[e] = __expf(acc[e] - mx); s += p[e]; }
        float inv = 1.f / s;
        int i1 = 0; float b1 = p[0];
#pragma unroll
        for (int e = 1; e < NEXP; ++e) if (p[e] > b1) { b1 = p[e]; i1 = e; }
        int i2 = -1; float b2 = -1.f;
#pragma unroll
        for (int e = 0; e < NEXP; ++e)
            if (e != i1 && p[e] > b2) { b2 = p[e]; i2 = e; }
        int pos1 = atomicAdd(&counts[i1], 1);
        tokList[i1 * NTOK + pos1] = tok;
        slotList[i1 * NTOK + pos1] = 0;
        gateList[i1 * NTOK + pos1] = b1 * inv;
        int pos2 = atomicAdd(&counts[i2], 1);
        tokList[i2 * NTOK + pos2] = tok;
        slotList[i2 * NTOK + pos2] = 1;
        gateList[i2 * NTOK + pos2] = b2 * inv;
    }
}

// ---------------------------------------------------------------- routed experts
// grid = (token-tiles of 32, experts); 256 threads = 8 waves.
// LDS: x tile (32 x DIMD bf16, padded) + h tile (32 x HID bf16, padded).
__global__ __launch_bounds__(256) void moe_routed(
        const float* __restrict__ x, const int* __restrict__ counts,
        const int* __restrict__ tokList, const int* __restrict__ slotList,
        const float* __restrict__ gateList,
        const unsigned short* __restrict__ w1b, const unsigned short* __restrict__ w3b,
        const unsigned short* __restrict__ w2b, float* __restrict__ part) {
    constexpr int XROW = DIMD + 8;   // bf16 elems; 16B-aligned rows
    constexpr int HROW = HID + 8;
    extern __shared__ unsigned short smem[];
    unsigned short* xbuf = smem;                 // MT x XROW
    unsigned short* hbuf = smem + MT * XROW;     // MT x HROW
    __shared__ int   s_tok[MT];
    __shared__ float s_gate[MT];
    __shared__ int   s_slot[MT];

    const int e = blockIdx.y;
    const int cnt = counts[e];
    const int base = blockIdx.x * MT;
    if (base >= cnt) return;                     // uniform early-out

    const int tid = threadIdx.x;
    if (tid < MT) {
        int idx = base + tid;
        if (idx < cnt) {
            s_tok[tid]  = tokList[e * NTOK + idx];
            s_gate[tid] = gateList[e * NTOK + idx];
            s_slot[tid] = slotList[e * NTOK + idx];
        } else { s_tok[tid] = -1; s_gate[tid] = 0.f; s_slot[tid] = 0; }
    }
    __syncthreads();

    // stage x tile as bf16 (zeros for invalid rows), float4 -> ushort4
    for (int i = tid; i < MT * (DIMD / 4); i += 256) {
        int m = i >> 9;                // / (DIMD/4)
        int c = (i & (DIMD / 4 - 1)) * 4;
        int tok = s_tok[m];
        float4 v = (tok >= 0) ? *(const float4*)(x + (size_t)tok * DIMD + c)
                              : make_float4(0.f, 0.f, 0.f, 0.f);
        ushort4 o;
        o.x = f2bf(v.x); o.y = f2bf(v.y); o.z = f2bf(v.z); o.w = f2bf(v.w);
        *(ushort4*)(xbuf + m * XROW + c) = o;
    }
    __syncthreads();

    const int wave = tid >> 5, lane = tid & 31;
    const int half = lane >> 4, nl = lane & 15;

    // phase 1: h = silu(x*w1^T) * (x*w3^T); 4 WMMAs per 2 B loads
    for (int ht = wave; ht < HID / 16; ht += 8) {
        v8f a1lo = {0.f,0.f,0.f,0.f,0.f,0.f,0.f,0.f}, a1hi = a1lo;
        v8f a3lo = a1lo, a3hi = a1lo;
        const unsigned short* w1r =
            w1b + ((size_t)e * HID + ht * 16 + nl) * DIMD + half * 16;
        const unsigned short* w3r =
            w3b + ((size_t)e * HID + ht * 16 + nl) * DIMD + half * 16;
        for (int k = 0; k < DIMD; k += 32) {
            v16bf aLo = load_a_lds(xbuf, XROW, k, lane);
            v16bf aHi = load_a_lds(xbuf + 16 * XROW, XROW, k, lane);
            v16bf b1 = load_b_glb(w1r + k);
            v16bf b3 = load_b_glb(w3r + k);
            a1lo = wmma_bf16(aLo, b1, a1lo);
            a1hi = wmma_bf16(aHi, b1, a1hi);
            a3lo = wmma_bf16(aLo, b3, a3lo);
            a3hi = wmma_bf16(aHi, b3, a3hi);
        }
#pragma unroll
        for (int r = 0; r < 8; ++r) {
            int m = r + half * 8;
            float v1 = a1lo[r], v3 = a3lo[r];
            hbuf[m * HROW + ht * 16 + nl] = f2bf((v1 / (1.f + __expf(-v1))) * v3);
            v1 = a1hi[r]; v3 = a3hi[r];
            hbuf[(16 + m) * HROW + ht * 16 + nl] =
                f2bf((v1 / (1.f + __expf(-v1))) * v3);
        }
    }
    __syncthreads();

    // phase 2: out = h * w2^T, gate-scaled, into per-slot partials
    for (int dt = wave; dt < DIMD / 16; dt += 8) {
        v8f accLo = {0.f,0.f,0.f,0.f,0.f,0.f,0.f,0.f}, accHi = accLo;
        const unsigned short* w2r =
            w2b + ((size_t)e * DIMD + dt * 16 + nl) * HID + half * 16;
        for (int k = 0; k < HID; k += 32) {
            v16bf b = load_b_glb(w2r + k);
            accLo = wmma_bf16(load_a_lds(hbuf, HROW, k, lane), b, accLo);
            accHi = wmma_bf16(load_a_lds(hbuf + 16 * HROW, HROW, k, lane), b, accHi);
        }
#pragma unroll
        for (int r = 0; r < 8; ++r) {
            int m = r + half * 8;
            int tok = s_tok[m];
            if (tok >= 0)
                part[((size_t)tok * 2 + s_slot[m]) * DIMD + dt * 16 + nl] =
                    s_gate[m] * accLo[r];
            tok = s_tok[16 + m];
            if (tok >= 0)
                part[((size_t)tok * 2 + s_slot[16 + m]) * DIMD + dt * 16 + nl] =
                    s_gate[16 + m] * accHi[r];
        }
    }
}

// ---------------------------------------------------------------- shared expert
__global__ __launch_bounds__(256) void moe_shared(
        const float* __restrict__ x,
        const unsigned short* __restrict__ sw1b, const unsigned short* __restrict__ sw3b,
        const unsigned short* __restrict__ sw2b, float* __restrict__ out) {
    constexpr int XROW = DIMD + 8;
    constexpr int SROW = HSH + 8;
    extern __shared__ unsigned short smem[];
    unsigned short* xbuf = smem;
    unsigned short* hbuf = smem + MT * XROW;

    const int tile = blockIdx.x;
    const int tid = threadIdx.x;

    for (int i = tid; i < MT * (DIMD / 4); i += 256) {
        int m = i >> 9;
        int c = (i & (DIMD / 4 - 1)) * 4;
        float4 v = *(const float4*)(x + (size_t)(tile * MT + m) * DIMD + c);
        ushort4 o;
        o.x = f2bf(v.x); o.y = f2bf(v.y); o.z = f2bf(v.z); o.w = f2bf(v.w);
        *(ushort4*)(xbuf + m * XROW + c) = o;
    }
    __syncthreads();

    const int wave = tid >> 5, lane = tid & 31;
    const int half = lane >> 4, nl = lane & 15;

    for (int ht = wave; ht < HSH / 16; ht += 8) {
        v8f a1lo = {0.f,0.f,0.f,0.f,0.f,0.f,0.f,0.f}, a1hi = a1lo;
        v8f a3lo = a1lo, a3hi = a1lo;
        const unsigned short* w1r = sw1b + (size_t)(ht * 16 + nl) * DIMD + half * 16;
        const unsigned short* w3r = sw3b + (size_t)(ht * 16 + nl) * DIMD + half * 16;
        for (int k = 0; k < DIMD; k += 32) {
            v16bf aLo = load_a_lds(xbuf, XROW, k, lane);
            v16bf aHi = load_a_lds(xbuf + 16 * XROW, XROW, k, lane);
            v16bf b1 = load_b_glb(w1r + k);
            v16bf b3 = load_b_glb(w3r + k);
            a1lo = wmma_bf16(aLo, b1, a1lo);
            a1hi = wmma_bf16(aHi, b1, a1hi);
            a3lo = wmma_bf16(aLo, b3, a3lo);
            a3hi = wmma_bf16(aHi, b3, a3hi);
        }
#pragma unroll
        for (int r = 0; r < 8; ++r) {
            int m = r + half * 8;
            float v1 = a1lo[r], v3 = a3lo[r];
            hbuf[m * SROW + ht * 16 + nl] = f2bf((v1 / (1.f + __expf(-v1))) * v3);
            v1 = a1hi[r]; v3 = a3hi[r];
            hbuf[(16 + m) * SROW + ht * 16 + nl] =
                f2bf((v1 / (1.f + __expf(-v1))) * v3);
        }
    }
    __syncthreads();

    for (int dt = wave; dt < DIMD / 16; dt += 8) {
        v8f accLo = {0.f,0.f,0.f,0.f,0.f,0.f,0.f,0.f}, accHi = accLo;
        const unsigned short* w2r = sw2b + (size_t)(dt * 16 + nl) * HSH + half * 16;
        for (int k = 0; k < HSH; k += 32) {
            v16bf b = load_b_glb(w2r + k);
            accLo = wmma_bf16(load_a_lds(hbuf, SROW, k, lane), b, accLo);
            accHi = wmma_bf16(load_a_lds(hbuf + 16 * SROW, SROW, k, lane), b, accHi);
        }
#pragma unroll
        for (int r = 0; r < 8; ++r) {
            int m = r + half * 8;
            out[(size_t)(tile * MT + m) * DIMD + dt * 16 + nl] = accLo[r];
            out[(size_t)(tile * MT + 16 + m) * DIMD + dt * 16 + nl] = accHi[r];
        }
    }
}

// ---------------------------------------------------------------- final reduce
__global__ void moe_final(float4* __restrict__ out, const float4* __restrict__ part,
                          long n4) {
    const long DC = DIMD / 4;
    long i = (long)blockIdx.x * blockDim.x + threadIdx.x;
    long stride = (long)gridDim.x * blockDim.x;
    for (; i < n4; i += stride) {
        long n = i / DC, c = i % DC;
        float4 o = out[i];
        float4 p0 = part[(n * 2) * DC + c];
        float4 p1 = part[(n * 2 + 1) * DC + c];
        o.x += p0.x + p1.x; o.y += p0.y + p1.y;
        o.z += p0.z + p1.z; o.w += p0.w + p1.w;
        out[i] = o;
    }
}

// ---------------------------------------------------------------- launch
extern "C" void kernel_launch(void* const* d_in, const int* in_sizes, int n_in,
                              void* d_out, int out_size, void* d_ws, size_t ws_size,
                              hipStream_t stream) {
    (void)in_sizes; (void)n_in; (void)out_size; (void)ws_size;
    const float* x   = (const float*)d_in[0];
    const float* gw  = (const float*)d_in[1];
    const float* w1  = (const float*)d_in[2];
    const float* w2  = (const float*)d_in[3];
    const float* w3  = (const float*)d_in[4];
    const float* sw1 = (const float*)d_in[5];
    const float* sw2 = (const float*)d_in[6];
    const float* sw3 = (const float*)d_in[7];
    float* out = (float*)d_out;

    char* ws = (char*)d_ws;
    size_t off = 0;
    auto take = [&](size_t bytes) {
        char* p = ws + off;
        off = (off + bytes + 255) & ~(size_t)255;
        return p;
    };
    int*   counts   = (int*)  take(NEXP * sizeof(int));
    int*   tokList  = (int*)  take((size_t)NEXP * NTOK * sizeof(int));
    int*   slotList = (int*)  take((size_t)NEXP * NTOK * sizeof(int));
    float* gateList = (float*)take((size_t)NEXP * NTOK * sizeof(float));
    float* part     = (float*)take((size_t)NTOK * 2 * DIMD * sizeof(float));
    unsigned short* w1b  = (unsigned short*)take((size_t)NEXP * HID * DIMD * 2);
    unsigned short* w3b  = (unsigned short*)take((size_t)NEXP * HID * DIMD * 2);
    unsigned short* w2b  = (unsigned short*)take((size_t)NEXP * DIMD * HID * 2);
    unsigned short* sw1b = (unsigned short*)take((size_t)HSH * DIMD * 2);
    unsigned short* sw3b = (unsigned short*)take((size_t)HSH * DIMD * 2);
    unsigned short* sw2b = (unsigned short*)take((size_t)DIMD * HSH * 2);

    moe_zero_counts<<<1, 32, 0, stream>>>(counts);

    const long nEW = (long)NEXP * HID * DIMD / 4;
    const long nSW = (long)HSH * DIMD / 4;
    moe_cvt_bf16<<<2048, 256, 0, stream>>>((const float4*)w1,  (ushort4*)w1b,  nEW);
    moe_cvt_bf16<<<2048, 256, 0, stream>>>((const float4*)w3,  (ushort4*)w3b,  nEW);
    moe_cvt_bf16<<<2048, 256, 0, stream>>>((const float4*)w2,  (ushort4*)w2b,  nEW);
    moe_cvt_bf16<<<1024, 256, 0, stream>>>((const float4*)sw1, (ushort4*)sw1b, nSW);
    moe_cvt_bf16<<<1024, 256, 0, stream>>>((const float4*)sw3, (ushort4*)sw3b, nSW);
    moe_cvt_bf16<<<1024, 256, 0, stream>>>((const float4*)sw2, (ushort4*)sw2b, nSW);

    moe_gate<<<NTOK / 8, 256, 0, stream>>>(x, gw, counts, tokList, slotList, gateList);

    const size_t ldsR = (size_t)MT * (DIMD + 8) * 2 + (size_t)MT * (HID + 8) * 2;
    moe_routed<<<dim3(NTOK / MT, NEXP), 256, ldsR, stream>>>(
        x, counts, tokList, slotList, gateList, w1b, w3b, w2b, part);

    const size_t ldsS = (size_t)MT * (DIMD + 8) * 2 + (size_t)MT * (HSH + 8) * 2;
    moe_shared<<<NTOK / MT, 256, ldsS, stream>>>(x, sw1b, sw3b, sw2b, out);

    moe_final<<<1024, 256, 0, stream>>>((float4*)out, (const float4*)part,
                                        (long)NTOK * DIMD / 4);
}
